// MultiHeadAttention_67585605370071
// MI455X (gfx1250) — matrix operand order repaired and checked by reading, not compiled, not run on hardware
//
#include <hip/hip_runtime.h>

typedef __attribute__((ext_vector_type(16))) __bf16 v16bf;
typedef __attribute__((ext_vector_type(8)))  __bf16 v8bf;
typedef __attribute__((ext_vector_type(2)))  __bf16 v2bf;
typedef __attribute__((ext_vector_type(8)))  float  v8f;
typedef int v4i __attribute__((vector_size(16)));

#define D_MODEL   1024
#define NUM_HEADS 16
#define DKH       64
#define SEQ       2048
#define BATCH     4
#define NTOK      (BATCH*SEQ)   // 8192 tokens
#define LDT       40            // LDS tile row stride (elements), pads banks, keeps 16B align

#define HAS_ASYNC_LDS __has_builtin(__builtin_amdgcn_global_load_async_to_lds_b128)

typedef __attribute__((address_space(1))) v4i* gv4p;   // global int4*
typedef __attribute__((address_space(3))) v4i* lv4p;   // LDS int4*

// native RNE f32 -> bf16 (should lower to v_cvt_pk_bf16_f32-class ops)
__device__ __forceinline__ unsigned int pack2(float a, float b) {
  v2bf p;
  p.x = (__bf16)a;
  p.y = (__bf16)b;
  return __builtin_bit_cast(unsigned int, p);
}

union FragA { v16bf v; v8bf h[2]; };

__device__ __forceinline__ v8f wmma_bf16(const FragA& a, const FragA& b, v8f c) {
  return __builtin_amdgcn_wmma_f32_16x16x32_bf16(false, a.v, false, b.v, (short)0, c, false, false);
}

// ---------------------------------------------------------------------------
// Kernel 1: Q/K/V projection.  out = bf16( x @ W^T + bias )
//   transpose_v==0 : out[(b*H+h)*S + s][d]   (row-major per head, d contiguous)
//   transpose_v==1 : out[(b*H+h)*64 + d][s]  (V transposed: s contiguous)
// Block tile: M=128 tokens x N=64 channels, K-chunks of 32, 8 waves.
// ---------------------------------------------------------------------------
__global__ __launch_bounds__(256)
void qkv_proj_kernel(const float* __restrict__ X, const float* __restrict__ W,
                     const float* __restrict__ bias, __bf16* __restrict__ out,
                     int transpose_v)
{
  __shared__ __align__(16) __bf16 ldsA[128 * LDT];
  __shared__ __align__(16) __bf16 ldsB[64 * LDT];

  const int tid  = threadIdx.x;
  const int wv   = tid >> 5;
  const int lane = tid & 31;
  const int hi   = lane >> 4;      // lane half
  const int l16  = lane & 15;
  const int mBase = blockIdx.x * 128;
  const int nBase = blockIdx.y * 64;

  v8f acc[4];
  for (int t = 0; t < 4; ++t)
    for (int j = 0; j < 8; ++j) acc[t][j] = 0.f;

  for (int kt = 0; kt < D_MODEL / 32; ++kt) {
    const int k0 = kt * 32;
    __syncthreads();
    { // stage X tile 128x32 f32 -> bf16
      const int r  = tid >> 1;
      const int c0 = (tid & 1) * 16;
      const float* src = X + (size_t)(mBase + r) * D_MODEL + k0 + c0;
      unsigned int* dst = (unsigned int*)&ldsA[r * LDT + c0];
#pragma unroll
      for (int j = 0; j < 4; ++j) {
        float4 f = *(const float4*)(src + j * 4);
        dst[j * 2 + 0] = pack2(f.x, f.y);
        dst[j * 2 + 1] = pack2(f.z, f.w);
      }
    }
    { // stage W tile 64x32 f32 -> bf16  (rows are output channels; W[o][d])
      const int r  = tid >> 2;
      const int c0 = (tid & 3) * 8;
      const float* src = W + (size_t)(nBase + r) * D_MODEL + k0 + c0;
      unsigned int* dst = (unsigned int*)&ldsB[r * LDT + c0];
#pragma unroll
      for (int j = 0; j < 2; ++j) {
        float4 f = *(const float4*)(src + j * 4);
        dst[j * 2 + 0] = pack2(f.x, f.y);
        dst[j * 2 + 1] = pack2(f.z, f.w);
      }
    }
    __syncthreads();

    // A fragment: row = wv*16 + l16; K pattern {0-7,16-23} / {8-15,24-31}
    FragA a;
    const __bf16* pa = &ldsA[(wv * 16 + l16) * LDT + hi * 8];
    a.h[0] = *(const v8bf*)(pa);
    a.h[1] = *(const v8bf*)(pa + 16);
#pragma unroll
    for (int nt = 0; nt < 4; ++nt) {
      // B fragment: col n = nt*16 + l16; K pattern {0-7,16-23} / {8-15,24-31}
      FragA b;
      const __bf16* pb = &ldsB[(nt * 16 + l16) * LDT + hi * 8];
      b.h[0] = *(const v8bf*)(pb);
      b.h[1] = *(const v8bf*)(pb + 16);
      acc[nt] = wmma_bf16(a, b, acc[nt]);
    }
  }

  // epilogue: bias add, bf16 convert, scatter to per-head layout
#pragma unroll
  for (int nt = 0; nt < 4; ++nt) {
    const int n  = nBase + nt * 16 + l16;
    const float bb = bias[n];
    const int hh = n >> 6, d = n & 63;
#pragma unroll
    for (int i = 0; i < 8; ++i) {
      const int m = mBase + wv * 16 + i + 8 * hi;   // token index
      const int b = m >> 11, s = m & (SEQ - 1);
      size_t idx;
      if (!transpose_v) idx = ((size_t)((b * NUM_HEADS + hh) * SEQ + s)) * DKH + d;
      else              idx = ((size_t)((b * NUM_HEADS + hh) * DKH + d)) * SEQ + s;
      out[idx] = (__bf16)(acc[nt][i] + bb);
    }
  }
}

// ---------------------------------------------------------------------------
// Kernel 2: flash-style attention. One wave = one (b,h,16-query) block.
// Q,K: [b*H+h][s][64] bf16.  Vt: [b*H+h][64][s] bf16.  Out: [token][1024] bf16.
// ---------------------------------------------------------------------------
__global__ __launch_bounds__(256)
void attention_kernel(const __bf16* __restrict__ Q,
                      const __bf16* __restrict__ K,
                      const __bf16* __restrict__ Vt,
                      __bf16* __restrict__ Obf)
{
  __shared__ __align__(16) __bf16 pbuf[8][16 * LDT];

  const int tid  = threadIdx.x;
  const int wv   = tid >> 5;
  const int lane = tid & 31;
  const int hi   = lane >> 4;
  const int l16  = lane & 15;
  __bf16* pw = pbuf[wv];

  const int task = blockIdx.x * 8 + wv;      // 0 .. 8191
  const int bh = task >> 7;                  // b*16 + h
  const int qb = task & 127;                 // query block within sequence

  const __bf16* qbase = Q  + ((size_t)bh * SEQ + qb * 16) * DKH;
  const __bf16* kbase = K  + (size_t)bh * SEQ * DKH;
  const __bf16* vbase = Vt + (size_t)bh * DKH * SEQ;

  // Q fragments (A operand), K-dim 64 = two 32-wide steps
  FragA qa[2];
  {
    const __bf16* qp = qbase + (size_t)l16 * DKH;
#pragma unroll
    for (int kk = 0; kk < 2; ++kk) {
      qa[kk].h[0] = *(const v8bf*)(qp + kk * 32 + hi * 8);
      qa[kk].h[1] = *(const v8bf*)(qp + kk * 32 + 16 + hi * 8);
    }
  }

  float mrow[8], lrow[8];
  v8f oacc[4];
  for (int i = 0; i < 8; ++i) { mrow[i] = -3.0e38f; lrow[i] = 0.f; }
  for (int t = 0; t < 4; ++t)
    for (int i = 0; i < 8; ++i) oacc[t][i] = 0.f;

  const float scale = 0.125f;  // 1/sqrt(64)

  for (int ch = 0; ch < SEQ / 32; ++ch) {
    // prefetch next chunk's K rows (32 rows -> one per lane) and V stream
    if (ch + 1 < SEQ / 32) {
      __builtin_prefetch(kbase + (size_t)((ch + 1) * 32 + lane) * DKH, 0, 3);
      __builtin_prefetch(vbase + (size_t)lane * SEQ + (ch + 1) * 32, 0, 3);
      __builtin_prefetch(vbase + (size_t)(32 + lane) * SEQ + (ch + 1) * 32, 0, 3);
    }

    // ---- scores: 16 queries x 32 keys ----
    v8f sc[2];
    for (int t = 0; t < 2; ++t)
      for (int i = 0; i < 8; ++i) sc[t][i] = 0.f;
#pragma unroll
    for (int nt = 0; nt < 2; ++nt) {
      const __bf16* kp = kbase + (size_t)(ch * 32 + nt * 16 + l16) * DKH;
#pragma unroll
      for (int kk = 0; kk < 2; ++kk) {
        FragA kf;   // B operand: lane = key col, d_k pattern {0-7,16-23} / {8-15,24-31}
        kf.h[0] = *(const v8bf*)(kp + kk * 32 + hi * 8);
        kf.h[1] = *(const v8bf*)(kp + kk * 32 + 16 + hi * 8);
        sc[nt] = wmma_bf16(qa[kk], kf, sc[nt]);
      }
    }

    // ---- online softmax (rows live in elem i + 8*hi; xor<16 stays in half) ----
    float mnew[8], alpha[8];
#pragma unroll
    for (int i = 0; i < 8; ++i) {
      float v = fmaxf(sc[0][i] * scale, sc[1][i] * scale);
      v = fmaxf(v, __shfl_xor(v, 1));
      v = fmaxf(v, __shfl_xor(v, 2));
      v = fmaxf(v, __shfl_xor(v, 4));
      v = fmaxf(v, __shfl_xor(v, 8));
      mnew[i]  = fmaxf(mrow[i], v);
      alpha[i] = __expf(mrow[i] - mnew[i]);
      mrow[i]  = mnew[i];
    }
#pragma unroll
    for (int i = 0; i < 8; ++i) {
      const float p0 = __expf(sc[0][i] * scale - mnew[i]);
      const float p1 = __expf(sc[1][i] * scale - mnew[i]);
      const int row = i + 8 * hi;               // C-layout row
      pw[row * LDT + l16]      = (__bf16)p0;
      pw[row * LDT + 16 + l16] = (__bf16)p1;
      float rs = p0 + p1;
      rs += __shfl_xor(rs, 1);
      rs += __shfl_xor(rs, 2);
      rs += __shfl_xor(rs, 4);
      rs += __shfl_xor(rs, 8);
      lrow[i] = lrow[i] * alpha[i] + rs;
    }
    asm volatile("s_wait_dscnt 0x0" ::: "memory");  // LDS stores -> lane-crossing reads

    // rescale running output
#pragma unroll
    for (int t = 0; t < 4; ++t)
#pragma unroll
      for (int i = 0; i < 8; ++i) oacc[t][i] *= alpha[i];

    // P fragment (A operand) from LDS: row = l16, A-layout K pattern
    FragA pf;
    const __bf16* pp = &pw[l16 * LDT + hi * 8];
    pf.h[0] = *(const v8bf*)(pp);
    pf.h[1] = *(const v8bf*)(pp + 16);

    // ---- P @ V : 4 output-dim subtiles ----
#pragma unroll
    for (int t = 0; t < 4; ++t) {
      const __bf16* vp = vbase + (size_t)(t * 16 + l16) * SEQ + ch * 32;
      FragA vf;   // B operand: lane = dim col, key pattern {0-7,16-23} / {8-15,24-31}
      vf.h[0] = *(const v8bf*)(vp + hi * 8);
      vf.h[1] = *(const v8bf*)(vp + 16 + hi * 8);
      oacc[t] = wmma_bf16(pf, vf, oacc[t]);
    }
  }

  // normalize + write merged-head bf16 activations [token][1024]
  const int b = bh >> 4, h = bh & 15;
#pragma unroll
  for (int t = 0; t < 4; ++t) {
    const int d = t * 16 + l16;
#pragma unroll
    for (int i = 0; i < 8; ++i) {
      const int s = qb * 16 + i + 8 * hi;
      const float val = oacc[t][i] / lrow[i];
      Obf[((size_t)(b * SEQ + s)) * D_MODEL + h * DKH + d] = (__bf16)val;
    }
  }
}

// ---------------------------------------------------------------------------
// Kernel 3: output projection.  d_out = f32( attn_bf @ Wo^T + bo )
// A-tile staged via async global->LDS DMA when the builtin is available.
// ---------------------------------------------------------------------------
__global__ __launch_bounds__(256)
void out_proj_kernel(const __bf16* __restrict__ Abf, const float* __restrict__ W,
                     const float* __restrict__ bias, float* __restrict__ out)
{
  __shared__ __align__(16) __bf16 ldsA[128 * LDT];
  __shared__ __align__(16) __bf16 ldsB[64 * LDT];

  const int tid  = threadIdx.x;
  const int wv   = tid >> 5;
  const int lane = tid & 31;
  const int hi   = lane >> 4;
  const int l16  = lane & 15;
  const int mBase = blockIdx.x * 128;
  const int nBase = blockIdx.y * 64;

  v8f acc[4];
  for (int t = 0; t < 4; ++t)
    for (int j = 0; j < 8; ++j) acc[t][j] = 0.f;

  for (int kt = 0; kt < D_MODEL / 32; ++kt) {
    const int k0 = kt * 32;
    __syncthreads();
    { // stage A tile 128x32 (already bf16): async DMA to LDS when available
      const int r  = tid >> 1;
      const int c0 = (tid & 1) * 16;
      const __bf16* src = Abf + (size_t)(mBase + r) * D_MODEL + k0 + c0;
      __bf16* dstg = &ldsA[r * LDT + c0];
#if HAS_ASYNC_LDS
      gv4p gsrc = (gv4p)src;
      lv4p ldst = (lv4p)dstg;
      __builtin_amdgcn_global_load_async_to_lds_b128(gsrc,     ldst,     0, 0);
      __builtin_amdgcn_global_load_async_to_lds_b128(gsrc + 1, ldst + 1, 0, 0);
#else
      uint4* dst = (uint4*)dstg;
      dst[0] = *(const uint4*)(src);
      dst[1] = *(const uint4*)(src + 8);
#endif
    }
    { // stage Wo tile 64x32 f32 -> bf16
      const int r  = tid >> 2;
      const int c0 = (tid & 3) * 8;
      const float* src = W + (size_t)(nBase + r) * D_MODEL + k0 + c0;
      unsigned int* dst = (unsigned int*)&ldsB[r * LDT + c0];
#pragma unroll
      for (int j = 0; j < 2; ++j) {
        float4 f = *(const float4*)(src + j * 4);
        dst[j * 2 + 0] = pack2(f.x, f.y);
        dst[j * 2 + 1] = pack2(f.z, f.w);
      }
    }
#if HAS_ASYNC_LDS
#if __has_builtin(__builtin_amdgcn_s_wait_asynccnt)
    __builtin_amdgcn_s_wait_asynccnt(0);
#else
    asm volatile("s_wait_asynccnt 0x0" ::: "memory");
#endif
#endif
    __syncthreads();

    FragA a;
    const __bf16* pa = &ldsA[(wv * 16 + l16) * LDT + hi * 8];
    a.h[0] = *(const v8bf*)(pa);
    a.h[1] = *(const v8bf*)(pa + 16);
#pragma unroll
    for (int nt = 0; nt < 4; ++nt) {
      FragA b;
      const __bf16* pb = &ldsB[(nt * 16 + l16) * LDT + hi * 8];
      b.h[0] = *(const v8bf*)(pb);
      b.h[1] = *(const v8bf*)(pb + 16);
      acc[nt] = wmma_bf16(a, b, acc[nt]);
    }
  }

#pragma unroll
  for (int nt = 0; nt < 4; ++nt) {
    const int n = nBase + nt * 16 + l16;
    const float bb = bias[n];
#pragma unroll
    for (int i = 0; i < 8; ++i) {
      const int m = mBase + wv * 16 + i + 8 * hi;
      out[(size_t)m * D_MODEL + n] = acc[nt][i] + bb;
    }
  }
}

// ---------------------------------------------------------------------------
extern "C" void kernel_launch(void* const* d_in, const int* in_sizes, int n_in,
                              void* d_out, int out_size, void* d_ws, size_t ws_size,
                              hipStream_t stream) {
  (void)in_sizes; (void)n_in; (void)out_size;
  const float* x  = (const float*)d_in[0];
  const float* Wq = (const float*)d_in[1];
  const float* bq = (const float*)d_in[2];
  const float* Wk = (const float*)d_in[3];
  const float* bk = (const float*)d_in[4];
  const float* Wv = (const float*)d_in[5];
  const float* bv = (const float*)d_in[6];
  const float* Wo = (const float*)d_in[7];
  const float* bo = (const float*)d_in[8];
  float* out = (float*)d_out;

  const size_t buf = (size_t)NTOK * D_MODEL;       // elements per bf16 buffer
  if (ws_size < buf * 4 * sizeof(__bf16)) return;
  __bf16* qbf = (__bf16*)d_ws;
  __bf16* kbf = qbf + buf;
  __bf16* vtb = kbf + buf;
  __bf16* abf = vtb + buf;

  dim3 gproj(NTOK / 128, D_MODEL / 64);
  qkv_proj_kernel<<<gproj, 256, 0, stream>>>(x, Wq, bq, qbf, 0);
  qkv_proj_kernel<<<gproj, 256, 0, stream>>>(x, Wk, bk, kbf, 0);
  qkv_proj_kernel<<<gproj, 256, 0, stream>>>(x, Wv, bv, vtb, 1);

  const int nTasks = BATCH * NUM_HEADS * (SEQ / 16);   // 8192 wave tasks
  attention_kernel<<<nTasks / 8, 256, 0, stream>>>(qbf, kbf, vtb, abf);

  out_proj_kernel<<<gproj, 256, 0, stream>>>(abf, Wo, bo, out);
}
